// VectorQuantizer_39943195853390
// MI455X (gfx1250) — matrix operand order repaired and checked by reading, not compiled, not run on hardware
//
#include <hip/hip_runtime.h>
#include <hip/hip_bf16.h>

typedef __attribute__((ext_vector_type(16))) __bf16 v16bf;
typedef __attribute__((ext_vector_type(8)))  float  v8f;
typedef int v4i __attribute__((vector_size(16)));
typedef unsigned long long u64;
typedef unsigned int       u32;
typedef unsigned short     u16;

// ---- problem constants (fixed by reference) ----
#define N_TOK    65536     // 32*2048 tokens
#define N_CODE   2048      // codebook entries
#define DIM      256       // embedding dim
#define Q_CNT    16777216  // N_TOK*DIM
#define IDX_OFF  16777217  // after quantized + loss
#define DIST_OFF 16842753  // after quantized + loss + indices

// tile config
#define TM 128
#define TN 128
#define TK 32
#define LDSS 40            // ushort stride (80B): 16B-aligned chunks, conflict-free

// ---- CDNA5 async global->LDS path (guarded; falls back to sync LDS copy) ----
#ifndef __has_builtin
#define __has_builtin(x) 0
#endif
#if __has_builtin(__builtin_amdgcn_global_load_async_to_lds_b128)
#define GEMM_ASYNC 1
#else
#define GEMM_ASYNC 0
#endif

#if GEMM_ASYNC
typedef __attribute__((address_space(1))) v4i* gv4i_p;  // global
typedef __attribute__((address_space(3))) v4i* lv4i_p;  // LDS
#endif

__device__ __forceinline__ void async_wait0() {
#if GEMM_ASYNC
#if __has_builtin(__builtin_amdgcn_s_wait_asynccnt)
  __builtin_amdgcn_s_wait_asynccnt(0);
#else
  asm volatile("s_wait_asynccnt 0" ::: "memory");
#endif
#endif
}

union Frag { v16bf v; uint4 q[2]; };

__device__ __forceinline__ void f32_split(float f, u16& hi, u16& lo) {
  u32 b = __float_as_uint(f);
  hi = (u16)(b >> 16);
  float fh = __uint_as_float(((u32)hi) << 16);
  float r = f - fh;
  lo = (u16)(__float_as_uint(r) >> 16);
}

// ---- init: argmin keys to +inf ----
__global__ void k_init(u64* __restrict__ minpack) {
  int i = blockIdx.x * blockDim.x + threadIdx.x;
  if (i < N_TOK) minpack[i] = ~0ull;
}

// ---- one-shot f32 -> bf16 hi/lo split (4 elems/thread) ----
__global__ __launch_bounds__(256) void k_split(const float* __restrict__ src,
                                               u16* __restrict__ hi,
                                               u16* __restrict__ lo, int n4) {
  int i = blockIdx.x * blockDim.x + threadIdx.x;
  if (i >= n4) return;
  float4 f = ((const float4*)src)[i];
  ushort4 h, l;
  f32_split(f.x, h.x, l.x);
  f32_split(f.y, h.y, l.y);
  f32_split(f.z, h.z, l.z);
  f32_split(f.w, h.w, l.w);
  ((ushort4*)hi)[i] = h;
  ((ushort4*)lo)[i] = l;
}

// ---- row squared norms (one wave per 256-float row) ----
__global__ void k_rowsq(const float* __restrict__ src, float* __restrict__ dst, int rows) {
  int wave = blockIdx.x * (blockDim.x >> 5) + (threadIdx.x >> 5);
  int lane = threadIdx.x & 31;
  if (wave >= rows) return;
  const float4* p = (const float4*)(src + (size_t)wave * DIM);
  float4 a = p[lane];
  float4 b = p[lane + 32];
  float s = a.x*a.x + a.y*a.y + a.z*a.z + a.w*a.w
          + b.x*b.x + b.y*b.y + b.z*b.z + b.w*b.w;
#pragma unroll
  for (int off = 16; off >= 1; off >>= 1) s += __shfl_xor(s, off, 32);
  if (lane == 0) dst[wave] = s;
}

// ---- fused distance GEMM: async double-buffered staging + bf16x3 WMMA + argmin ----
__global__ __launch_bounds__(256) void k_vq_gemm(
    const u16* __restrict__ Xh, const u16* __restrict__ Xl,
    const u16* __restrict__ Eh, const u16* __restrict__ El,
    const float* __restrict__ xsq, const float* __restrict__ esq,
    u64* __restrict__ minpack, float* __restrict__ distOut)
{
  // [buffer][array: Xh,Xl,Eh,El][row][k] — 81920 B
  __shared__ __align__(16) u16 smem[2][4][TM][LDSS];

  const int n0 = blockIdx.x * TN;
  const int m0 = blockIdx.y * TM;
  const int tid  = threadIdx.x;
  const int lane = tid & 31;
  const int wave = tid >> 5;
  const int wr = (wave & 1) * 64;   // wave row offset within tile
  const int wc = (wave >> 1) * 32;  // wave col offset within tile

  const u16* bases[4] = { Xh + (size_t)m0 * DIM, Xl + (size_t)m0 * DIM,
                          Eh + (size_t)n0 * DIM, El + (size_t)n0 * DIM };

  // stage one 4x(128x32) bf16 slab: 2048 x 16B chunks, 8 per thread
  auto stage = [&](int nb, int kc) {
#pragma unroll
    for (int j = 0; j < 8; ++j) {
      int lin = j * 256 + tid;          // wave-uniform arr (512 chunks/array)
      int arr = lin >> 9;
      int cid = lin & 511;
      int row = cid >> 2;
      int ch  = (cid & 3) * 8;          // u16 elements, 16B chunks
      const u16* gsrc = bases[arr] + (size_t)row * DIM + kc + ch;
      u16* ldst = &smem[nb][arr][row][ch];
#if GEMM_ASYNC
      __builtin_amdgcn_global_load_async_to_lds_b128((gv4i_p)gsrc, (lv4i_p)ldst, 0, 0);
#else
      *(uint4*)ldst = *(const uint4*)gsrc;
#endif
    }
  };

  const v8f vzero = {0.f,0.f,0.f,0.f,0.f,0.f,0.f,0.f};
  v8f acc[4][2];
#pragma unroll
  for (int r = 0; r < 4; ++r)
#pragma unroll
    for (int c = 0; c < 2; ++c) acc[r][c] = vzero;

  // prologue: fill buffer 0
  stage(0, 0);
  async_wait0();
  __syncthreads();

  for (int ic = 0; ic < DIM / TK; ++ic) {
    const int cur = ic & 1;
    if (ic + 1 < DIM / TK) stage(cur ^ 1, (ic + 1) * TK);  // overlap next slab

    // ---- load fragments per ISA bf16 layouts ----
    Frag ah[4], al[4], bh[2], bl[2];
    const int kA   = (lane >> 4) * 8;   // A: lanes<16 -> K{0..7,16..23}, else {8..15,24..31}
    const int rowA = wr + (lane & 15);
#pragma unroll
    for (int rt = 0; rt < 4; ++rt) {
      const u16* ph = &smem[cur][0][rowA + rt * 16][kA];
      const u16* pl = &smem[cur][1][rowA + rt * 16][kA];
      ah[rt].q[0] = *(const uint4*)(ph);
      ah[rt].q[1] = *(const uint4*)(ph + 16);
      al[rt].q[0] = *(const uint4*)(pl);
      al[rt].q[1] = *(const uint4*)(pl + 16);
    }
    const int kB   = (lane >> 4) * 8;   // B: lanes<16 -> K{0..7,16..23}, else {8..15,24..31}
    const int rowB = wc + (lane & 15);
#pragma unroll
    for (int ct = 0; ct < 2; ++ct) {
      const u16* ph = &smem[cur][2][rowB + ct * 16][kB];
      const u16* pl = &smem[cur][3][rowB + ct * 16][kB];
      bh[ct].q[0] = *(const uint4*)(ph);
      bh[ct].q[1] = *(const uint4*)(ph + 16);
      bl[ct].q[0] = *(const uint4*)(pl);
      bl[ct].q[1] = *(const uint4*)(pl + 16);
    }

    // ---- split-precision MACs: hi*hi + hi*lo + lo*hi ----
#pragma unroll
    for (int rt = 0; rt < 4; ++rt) {
#pragma unroll
      for (int ct = 0; ct < 2; ++ct) {
        acc[rt][ct] = __builtin_amdgcn_wmma_f32_16x16x32_bf16(
            false, ah[rt].v, false, bh[ct].v, (short)0, acc[rt][ct], false, false);
        acc[rt][ct] = __builtin_amdgcn_wmma_f32_16x16x32_bf16(
            false, ah[rt].v, false, bl[ct].v, (short)0, acc[rt][ct], false, false);
        acc[rt][ct] = __builtin_amdgcn_wmma_f32_16x16x32_bf16(
            false, al[rt].v, false, bh[ct].v, (short)0, acc[rt][ct], false, false);
      }
    }
    async_wait0();      // my slab copies landed
    __syncthreads();    // everyone's landed; everyone done reading cur
  }

  // ---- epilogue: distances + fused row-argmin (packed u64 min) ----
  const int colBase = n0 + wc + (lane & 15);
  const int rowHalf = (lane >> 4) * 8;   // C/D layout: VGPR v -> row v (lanes<16) or v+8
#pragma unroll
  for (int rt = 0; rt < 4; ++rt) {
#pragma unroll
    for (int v = 0; v < 8; ++v) {
      const int m = m0 + wr + rt * 16 + rowHalf + v;
      const float xs = xsq[m];
      u64 best = ~0ull;
#pragma unroll
      for (int ct = 0; ct < 2; ++ct) {
        const int n = colBase + ct * 16;
        float d = xs + esq[n] - 2.0f * acc[rt][ct][v];
        distOut[(size_t)m * N_CODE + n] = d;
        u64 key = ((u64)__float_as_uint(d) << 32) | (u32)n;  // d>=0: bit order == value order
        best = (key < best) ? key : best;
      }
#pragma unroll
      for (int off = 8; off >= 1; off >>= 1) {  // reduce across the 16-lane half
        u32 lo = (u32)best, hi = (u32)(best >> 32);
        u32 olo = __shfl_xor(lo, off, 32);
        u32 ohi = __shfl_xor(hi, off, 32);
        u64 other = ((u64)ohi << 32) | olo;
        best = (other < best) ? other : best;
      }
      if ((lane & 15) == 0) atomicMin(&minpack[m], best);
    }
  }
}

// ---- gather winners, write quantized + indices, deterministic partial loss ----
__global__ __launch_bounds__(256) void k_finalize(
    const float* __restrict__ X, const float* __restrict__ E,
    const u64* __restrict__ minpack, float* __restrict__ qout,
    int* __restrict__ idxout, float* __restrict__ partials)
{
  __shared__ float wsum[8];
  const int token = blockIdx.x * 4 + (threadIdx.x >> 6);
  const int sub   = threadIdx.x & 63;
  const u32 idx = (u32)(minpack[token] & 0xffffffffull);
  const float4 q = ((const float4*)(E + (size_t)idx   * DIM))[sub];
  const float4 x = ((const float4*)(X + (size_t)token * DIM))[sub];
  ((float4*)(qout + (size_t)token * DIM))[sub] = q;  // straight-through fwd == quantized
  if (sub == 0) idxout[token] = (int)idx;
  float dx = q.x - x.x, dy = q.y - x.y, dz = q.z - x.z, dw = q.w - x.w;
  float s = dx*dx + dy*dy + dz*dz + dw*dw;
#pragma unroll
  for (int off = 16; off >= 1; off >>= 1) s += __shfl_xor(s, off, 32);
  if ((threadIdx.x & 31) == 0) wsum[threadIdx.x >> 5] = s;
  __syncthreads();
  if (threadIdx.x == 0) {
    float t = 0.f;
#pragma unroll
    for (int w = 0; w < 8; ++w) t += wsum[w];  // fixed order -> deterministic
    partials[blockIdx.x] = t;
  }
}

// ---- deterministic final loss reduction ----
__global__ __launch_bounds__(256) void k_writeloss(const float* __restrict__ partials,
                                                   float* __restrict__ lossOut)
{
  __shared__ float wsum[8];
  float s = 0.f;
  for (int i = threadIdx.x; i < 16384; i += 256) s += partials[i];
#pragma unroll
  for (int off = 16; off >= 1; off >>= 1) s += __shfl_xor(s, off, 32);
  if ((threadIdx.x & 31) == 0) wsum[threadIdx.x >> 5] = s;
  __syncthreads();
  if (threadIdx.x == 0) {
    float t = 0.f;
#pragma unroll
    for (int w = 0; w < 8; ++w) t += wsum[w];
    // loss = q_latent + 0.25*e_latent; both equal mean((q-x)^2) numerically
    lossOut[0] = 1.25f * t / (float)Q_CNT;
  }
}

extern "C" void kernel_launch(void* const* d_in, const int* in_sizes, int n_in,
                              void* d_out, int out_size, void* d_ws, size_t ws_size,
                              hipStream_t stream) {
  (void)in_sizes; (void)n_in; (void)out_size; (void)ws_size;
  const float* X = (const float*)d_in[0];   // [65536, 256]
  const float* E = (const float*)d_in[1];   // [2048, 256]

  float* out      = (float*)d_out;
  float* qout     = out;                    // [65536,256]
  float* loss_out = out + Q_CNT;            // [1]
  int*   idx_out  = (int*)(out + IDX_OFF);  // [65536] int32 bits
  float* dist_out = out + DIST_OFF;         // [65536,2048]

  char* ws = (char*)d_ws;
  u64*   minpack  = (u64*)ws;                   //   0 .. 512K
  float* xsq      = (float*)(ws + 524288);      // 512K .. 768K
  float* esq      = (float*)(ws + 786432);      // 768K .. 776K
  float* partials = (float*)(ws + 794624);      // 776K .. 840K
  u16*   Xh = (u16*)(ws + (1ull << 20));                    // 32 MB
  u16*   Xl = (u16*)(ws + (1ull << 20) + 33554432ull);      // 32 MB
  u16*   Eh = (u16*)(ws + (1ull << 20) + 67108864ull);      //  1 MB
  u16*   El = (u16*)(ws + (1ull << 20) + 68157440ull);      //  1 MB

  k_init    <<<256,   256, 0, stream>>>(minpack);
  k_split   <<<16384, 256, 0, stream>>>(X, Xh, Xl, Q_CNT / 4);
  k_split   <<<512,   256, 0, stream>>>(E, Eh, El, (N_CODE * DIM) / 4);
  k_rowsq   <<<8192,  256, 0, stream>>>(X, xsq, N_TOK);
  k_rowsq   <<<256,   256, 0, stream>>>(E, esq, N_CODE);
  dim3 g(N_CODE / TN, N_TOK / TM);              // (16, 512)
  k_vq_gemm <<<g,     256, 0, stream>>>(Xh, Xl, Eh, El, xsq, esq, minpack, dist_out);
  k_finalize<<<16384, 256, 0, stream>>>(X, E, minpack, qout, idx_out, partials);
  k_writeloss<<<1,    256, 0, stream>>>(partials, loss_out);
}
